// CrissCrossAttention_7000796692591
// MI455X (gfx1250) — matrix operand order repaired and checked by reading, not compiled, not run on hardware
//
#include <hip/hip_runtime.h>

// ---------------- problem constants ----------------
constexpr int B_    = 4;
constexpr int C_    = 256;
constexpr int H_    = 96;
constexpr int W_    = 96;
constexpr int HEADS = 8;
constexpr int HD    = C_ / HEADS;     // 32
constexpr int NH    = B_ * HEADS;     // 32
constexpr int HW    = H_ * W_;        // 9216

typedef __attribute__((ext_vector_type(16))) _Float16 v16h;
typedef __attribute__((ext_vector_type(8)))  float    v8f;

// WMMA 16x16x32 f16 fragment index maps (CDNA5 ISA 7.12.2, wave32)
//   A (16x32, MxK): lane half h=lane>>4, row M=lane&15, elem e -> K = 8h + e (+8 if e>=8)
//   B (32x16, KxN): col N=lane&15, elem e -> K = 8h + e (+8 if e>=8)
//   C/D (16x16):    VGPR r -> M = r + 8h, N = lane&15
__device__ __forceinline__ int a_kmap(int h, int e) { return 8 * h + e + ((e >= 8) ? 8 : 0); }
__device__ __forceinline__ int b_kmap(int h, int e) { return 8 * h + e + ((e >= 8) ? 8 : 0); }

// ============================================================
// Kernel 1: fused QKV projection GEMM (fp32 in, f16 out, fp32 accum)
//   out[b, o, hw] = sum_c W[o,c] * x[b,c,hw] + bias[o]
//   Each wave computes a 16(M) x 32(N) strip: one A fragment feeds two
//   back-to-back WMMAs (A reuse + XDL pipelining).
//   grid: (HW/128, C/16, B*3), block: 128 (4 waves)
// ============================================================
__global__ __launch_bounds__(128) void qkv_proj(
    const float* __restrict__ x,
    const float* __restrict__ Wq, const float* __restrict__ bq,
    const float* __restrict__ Wk, const float* __restrict__ bk,
    const float* __restrict__ Wv, const float* __restrict__ bv,
    _Float16* __restrict__ q16, _Float16* __restrict__ k16, _Float16* __restrict__ v16)
{
    const int tid  = threadIdx.x;
    const int wave = tid >> 5;
    const int lane = tid & 31;
    const int h    = lane >> 4;
    const int ln   = lane & 15;

    const int n0 = blockIdx.x * 128 + wave * 32;  // hw strip start (32 wide)
    const int m0 = blockIdx.y * 16;               // out-channel tile start
    const int b  = blockIdx.z / 3;
    const int sel = blockIdx.z % 3;

    const float*  Wm   = (sel == 0) ? Wq : (sel == 1) ? Wk : Wv;
    const float*  bias = (sel == 0) ? bq : (sel == 1) ? bk : bv;
    _Float16*     outp = (sel == 0) ? q16 : (sel == 1) ? k16 : v16;

    const float* xb = x + (size_t)b * C_ * HW;

    v8f acc0 = {}, acc1 = {};
    for (int k0 = 0; k0 < C_; k0 += 32) {
        v16h a, bf0, bf1;
        // A = W[o, c]: contiguous in c
        const float* ap = Wm + (size_t)(m0 + ln) * C_ + k0 + 8 * h;
#pragma unroll
        for (int e = 0; e < 8; ++e)  a[e] = (_Float16)ap[e];
#pragma unroll
        for (int e = 8; e < 16; ++e) a[e] = (_Float16)ap[e + 8];
        // B = x[c, hw]: stride HW in k; two adjacent 16-wide column tiles
#pragma unroll
        for (int e = 0; e < 16; ++e) {
            const int kb = k0 + b_kmap(h, e);
            const float* bp = xb + (size_t)kb * HW + n0 + ln;
            bf0[e] = (_Float16)bp[0];
            bf1[e] = (_Float16)bp[16];
        }
        // Prefetch next K-slab (x is L2-resident; hide L2->WGP latency)
        if (k0 + 32 < C_) {
            __builtin_prefetch(ap + 32, 0, 3);
            __builtin_prefetch(xb + (size_t)(k0 + 32 + 16 * h) * HW + n0 + ln, 0, 3);
        }
        acc0 = __builtin_amdgcn_wmma_f32_16x16x32_f16(false, a, false, bf0,
                                                      (short)0, acc0, false, false);
        acc1 = __builtin_amdgcn_wmma_f32_16x16x32_f16(false, a, false, bf1,
                                                      (short)0, acc1, false, false);
    }
#pragma unroll
    for (int r = 0; r < 8; ++r) {
        const int mm = m0 + r + 8 * h;
        const float bb = bias[mm];
        const size_t base = (size_t)(b * C_ + mm) * HW + n0 + ln;
        outp[base]      = (_Float16)(acc0[r] + bb);
        outp[base + 16] = (_Float16)(acc1[r] + bb);
    }
}

// ============================================================
// Kernel 2: row-attention pass, one block per (n, y).
//   E[x,i] = sum_c q[c,x] k[c,i];  m,s = online softmax stats over i
//   O_row_un[c,x] = sum_i exp(E[x,i]-m[x]) v[c,i]
// ============================================================
__global__ __launch_bounds__(128) void row_pass(
    const _Float16* __restrict__ q16, const _Float16* __restrict__ k16,
    const _Float16* __restrict__ v16,
    float* __restrict__ o_row, float* __restrict__ m_row, float* __restrict__ s_row)
{
    const int y   = blockIdx.x;
    const int nh  = blockIdx.y;
    const int tid = threadIdx.x;
    const int wave = tid >> 5, lane = tid & 31;
    const int h = lane >> 4, ln = lane & 15;

    __shared__ _Float16 sQ[HD * W_];
    __shared__ _Float16 sK[HD * W_];
    __shared__ _Float16 sV[HD * W_];
    __shared__ float    sE[W_ * W_];
    _Float16* sPh = (_Float16*)sE;          // P overlaid on E row-by-row (byte-safe)
    constexpr int PROW = 2 * W_;            // half-stride so row x overlays float row x

    // ---- stage q/k/v row slices: sX[c*W + x] ----
    for (int i = tid; i < HD * W_; i += 128) {
        const int c = i / W_, xx = i - c * W_;
        const size_t g = (size_t)(nh * HD + c) * HW + (size_t)y * W_ + xx;
        sQ[i] = q16[g]; sK[i] = k16[g]; sV[i] = v16[g];
    }
    __syncthreads();

    // ---- E = q^T k : M=96(x), N=96(i), K=32, 36 tiles, 1 wmma each ----
    for (int t = wave; t < 36; t += 4) {
        const int X0 = (t / 6) * 16, I0 = (t % 6) * 16;
        v16h a, bf;
#pragma unroll
        for (int e = 0; e < 16; ++e) {
            a[e]  = sQ[a_kmap(h, e) * W_ + X0 + ln];
            bf[e] = sK[b_kmap(h, e) * W_ + I0 + ln];
        }
        v8f acc = {};
        acc = __builtin_amdgcn_wmma_f32_16x16x32_f16(false, a, false, bf,
                                                     (short)0, acc, false, false);
#pragma unroll
        for (int r = 0; r < 8; ++r)
            sE[(X0 + r + 8 * h) * W_ + I0 + ln] = acc[r];
    }
    __syncthreads();

    // ---- row softmax stats + P = exp(E - m) as f16 (in place) ----
    if (tid < W_) {
        const int xx = tid;
        float m = -1e30f;
        for (int i = 0; i < W_; ++i) m = fmaxf(m, sE[xx * W_ + i]);
        float s = 0.f;
        for (int i = 0; i < W_; ++i) {
            const float p = __expf(sE[xx * W_ + i] - m);
            s += p;
            sPh[xx * PROW + i] = (_Float16)p;
        }
        const size_t sidx = (size_t)nh * HW + (size_t)y * W_ + xx;
        m_row[sidx] = m;
        s_row[sidx] = s;
    }
    __syncthreads();

    // ---- O = P * V^T : M=96(x), N=32(c), K=96(i) ----
    for (int t = wave; t < 12; t += 4) {
        const int X0 = (t / 2) * 16, C0 = (t % 2) * 16;
        v8f acc = {};
        for (int k0 = 0; k0 < W_; k0 += 32) {
            v16h a, bf;
#pragma unroll
            for (int e = 0; e < 16; ++e) {
                a[e]  = sPh[(X0 + ln) * PROW + k0 + a_kmap(h, e)];
                bf[e] = sV[(C0 + ln) * W_ + k0 + b_kmap(h, e)];
            }
            acc = __builtin_amdgcn_wmma_f32_16x16x32_f16(false, a, false, bf,
                                                         (short)0, acc, false, false);
        }
#pragma unroll
        for (int r = 0; r < 8; ++r) {
            const int xx = X0 + r + 8 * h, cc = C0 + ln;
            o_row[(size_t)(nh * HD + cc) * HW + (size_t)y * W_ + xx] = acc[r];
        }
    }
}

// ============================================================
// Kernel 3: column-attention pass, one block per (n, x).
//   E[y,z] = sum_c q[c,y] k[c,z], diag(y==z) masked to -1e9
// ============================================================
__global__ __launch_bounds__(128) void col_pass(
    const _Float16* __restrict__ q16, const _Float16* __restrict__ k16,
    const _Float16* __restrict__ v16,
    float* __restrict__ o_col, float* __restrict__ m_col, float* __restrict__ s_col)
{
    const int xc  = blockIdx.x;
    const int nh  = blockIdx.y;
    const int tid = threadIdx.x;
    const int wave = tid >> 5, lane = tid & 31;
    const int h = lane >> 4, ln = lane & 15;

    __shared__ _Float16 sQ[HD * H_];
    __shared__ _Float16 sK[HD * H_];
    __shared__ _Float16 sV[HD * H_];
    __shared__ float    sE[H_ * H_];
    _Float16* sPh = (_Float16*)sE;
    constexpr int PROW = 2 * H_;

    // ---- stage q/k/v column slices: sX[c*H + y] ----
    for (int i = tid; i < HD * H_; i += 128) {
        const int c = i / H_, yy = i - c * H_;
        const size_t g = (size_t)(nh * HD + c) * HW + (size_t)yy * W_ + xc;
        sQ[i] = q16[g]; sK[i] = k16[g]; sV[i] = v16[g];
    }
    __syncthreads();

    // ---- E = q^T k over the column, with diagonal mask ----
    for (int t = wave; t < 36; t += 4) {
        const int Y0 = (t / 6) * 16, Z0 = (t % 6) * 16;
        v16h a, bf;
#pragma unroll
        for (int e = 0; e < 16; ++e) {
            a[e]  = sQ[a_kmap(h, e) * H_ + Y0 + ln];
            bf[e] = sK[b_kmap(h, e) * H_ + Z0 + ln];
        }
        v8f acc = {};
        acc = __builtin_amdgcn_wmma_f32_16x16x32_f16(false, a, false, bf,
                                                     (short)0, acc, false, false);
#pragma unroll
        for (int r = 0; r < 8; ++r) {
            const int yy = Y0 + r + 8 * h, zz = Z0 + ln;
            sE[yy * H_ + zz] = (yy == zz) ? -1e9f : acc[r];
        }
    }
    __syncthreads();

    // ---- column softmax stats + P ----
    if (tid < H_) {
        const int yy = tid;
        float m = -1e30f;
        for (int z = 0; z < H_; ++z) m = fmaxf(m, sE[yy * H_ + z]);
        float s = 0.f;
        for (int z = 0; z < H_; ++z) {
            const float p = __expf(sE[yy * H_ + z] - m);
            s += p;
            sPh[yy * PROW + z] = (_Float16)p;
        }
        const size_t sidx = (size_t)nh * HW + (size_t)yy * W_ + xc;
        m_col[sidx] = m;
        s_col[sidx] = s;
    }
    __syncthreads();

    // ---- O = P * V^T : M=96(y), N=32(c), K=96(z) ----
    for (int t = wave; t < 12; t += 4) {
        const int Y0 = (t / 2) * 16, C0 = (t % 2) * 16;
        v8f acc = {};
        for (int k0 = 0; k0 < H_; k0 += 32) {
            v16h a, bf;
#pragma unroll
            for (int e = 0; e < 16; ++e) {
                a[e]  = sPh[(Y0 + ln) * PROW + k0 + a_kmap(h, e)];
                bf[e] = sV[(C0 + ln) * H_ + k0 + b_kmap(h, e)];
            }
            acc = __builtin_amdgcn_wmma_f32_16x16x32_f16(false, a, false, bf,
                                                         (short)0, acc, false, false);
        }
#pragma unroll
        for (int r = 0; r < 8; ++r) {
            const int yy = Y0 + r + 8 * h, cc = C0 + ln;
            o_col[(size_t)(nh * HD + cc) * HW + (size_t)yy * W_ + xc] = acc[r];
        }
    }
}

// ============================================================
// Kernel 4: exact softmax recombination + gamma*out + x
// ============================================================
__global__ void combine(
    const float* __restrict__ xin, const float* __restrict__ gamma,
    const float* __restrict__ o_row, const float* __restrict__ o_col,
    const float* __restrict__ m_row, const float* __restrict__ s_row,
    const float* __restrict__ m_col, const float* __restrict__ s_col,
    float* __restrict__ out, int total)
{
    const int idx = blockIdx.x * blockDim.x + threadIdx.x;
    if (idx >= total) return;
    const int ch  = idx / HW;
    const int rem = idx - ch * HW;
    const int nh  = ch >> 5;               // / HD
    const int sidx = nh * HW + rem;

    const float mr = m_row[sidx], sr = s_row[sidx];
    const float mc = m_col[sidx], sc = s_col[sidx];
    const float m  = fmaxf(mr, mc);
    const float er = __expf(mr - m), ec = __expf(mc - m);
    const float denom = sr * er + sc * ec;
    const float o = (o_row[idx] * er + o_col[idx] * ec) / denom;
    out[idx] = gamma[0] * o + xin[idx];
}

// ============================================================
extern "C" void kernel_launch(void* const* d_in, const int* in_sizes, int n_in,
                              void* d_out, int out_size, void* d_ws, size_t ws_size,
                              hipStream_t stream)
{
    (void)in_sizes; (void)n_in; (void)out_size; (void)ws_size;

    const float* x     = (const float*)d_in[0];
    const float* Wq    = (const float*)d_in[1];
    const float* bq    = (const float*)d_in[2];
    const float* Wk    = (const float*)d_in[3];
    const float* bk    = (const float*)d_in[4];
    const float* Wv    = (const float*)d_in[5];
    const float* bv    = (const float*)d_in[6];
    const float* gamma = (const float*)d_in[7];
    float* out = (float*)d_out;

    constexpr size_t QKV_ELEMS = (size_t)B_ * C_ * HW;      // 9,437,184
    constexpr size_t QKV_BYTES = QKV_ELEMS * sizeof(_Float16);
    constexpr size_t O_BYTES   = QKV_ELEMS * sizeof(float);
    constexpr size_t S_BYTES   = (size_t)NH * HW * sizeof(float);

    char*  ws = (char*)d_ws;
    size_t off = 0;
    auto alloc = [&](size_t bytes) {
        void* p = ws + off;
        off = (off + bytes + 255) & ~(size_t)255;
        return p;
    };
    _Float16* q16   = (_Float16*)alloc(QKV_BYTES);
    _Float16* k16   = (_Float16*)alloc(QKV_BYTES);
    _Float16* v16   = (_Float16*)alloc(QKV_BYTES);
    float*    o_row = (float*)alloc(O_BYTES);
    float*    o_col = (float*)alloc(O_BYTES);
    float*    m_row = (float*)alloc(S_BYTES);
    float*    s_row = (float*)alloc(S_BYTES);
    float*    m_col = (float*)alloc(S_BYTES);
    float*    s_col = (float*)alloc(S_BYTES);

    // 1) QKV projection: grid (HW/128, C/16, B*3)
    qkv_proj<<<dim3(HW / 128, C_ / 16, B_ * 3), 128, 0, stream>>>(
        x, Wq, bq, Wk, bk, Wv, bv, q16, k16, v16);

    // 2) row attention pass: one block per (y, nh)
    row_pass<<<dim3(H_, NH), 128, 0, stream>>>(q16, k16, v16, o_row, m_row, s_row);

    // 3) column attention pass: one block per (x, nh)
    col_pass<<<dim3(W_, NH), 128, 0, stream>>>(q16, k16, v16, o_col, m_col, s_col);

    // 4) combine + residual
    const int total = (int)QKV_ELEMS;
    combine<<<(total + 255) / 256, 256, 0, stream>>>(
        x, gamma, o_row, o_col, m_row, s_row, m_col, s_col, out, total);
}
